// MoEBlock_17489106829865
// MI455X (gfx1250) — matrix operand order repaired and checked by reading, not compiled, not run on hardware
//
#include <hip/hip_runtime.h>

#define H    1024
#define F    4096
#define FH   2048   // F/2: each workgroup handles one half of the hidden dim
#define E    8
#define TT   8192   // total tokens (4 * 2048)
#define TK   2      // top-k
#define ROWS 32     // routed tokens per workgroup (2 WMMA m-tiles)

typedef __attribute__((ext_vector_type(16))) __bf16 bf16x16;
typedef __attribute__((ext_vector_type(8)))  __bf16 bf16x8;
typedef __attribute__((ext_vector_type(4)))  __bf16 bf16x4;
typedef __attribute__((ext_vector_type(8)))  float  f32x8;

static __device__ __forceinline__ f32x8 wmma_bf16(bf16x16 a, bf16x16 b, f32x8 c) {
    return __builtin_amdgcn_wmma_f32_16x16x32_bf16(false, a, false, b, (short)0, c, false, false);
}

// Build a 16x32 bf16 A-fragment (contiguous k order, k = 16h + s) from an LDS row.
// lanes 0-15 hold K = {k0..k0+15}; lanes 16-31 take +16.
static __device__ __forceinline__ bf16x16 frag_a(const __bf16* row, int k0, int off) {
    bf16x8 lo = *(const bf16x8*)(row + k0 + off);
    bf16x8 hi = *(const bf16x8*)(row + k0 + 8 + off);
    bf16x16 a;
#pragma unroll
    for (int i = 0; i < 8; ++i) { a[i] = lo[i]; a[i + 8] = hi[i]; }
    return a;
}

// ---------------------------------------------------------------- zero / init
__global__ __launch_bounds__(256) void k_zero(float* __restrict__ out,
                                              int* __restrict__ cnt,
                                              float* __restrict__ probsum) {
    int stride = gridDim.x * blockDim.x;
    for (int i = blockIdx.x * blockDim.x + threadIdx.x; i < TT * H; i += stride)
        out[i] = 0.0f;
    if (blockIdx.x == 0 && threadIdx.x < E) {
        cnt[threadIdx.x] = 0;
        probsum[threadIdx.x] = 0.0f;
    }
}

// ------------------------------------------------------- fp32 -> bf16 weights
__global__ __launch_bounds__(256) void k_convert(const float* __restrict__ up,
                                                 const float* __restrict__ down,
                                                 __bf16* __restrict__ upb,
                                                 __bf16* __restrict__ downb) {
    const int N4 = (E * F * H) / 4;
    int stride = gridDim.x * blockDim.x;
    for (int i = blockIdx.x * blockDim.x + threadIdx.x; i < N4; i += stride) {
        float4 u = ((const float4*)up)[i];
        bf16x4 a;
        a[0] = (__bf16)u.x; a[1] = (__bf16)u.y; a[2] = (__bf16)u.z; a[3] = (__bf16)u.w;
        ((bf16x4*)upb)[i] = a;
        float4 d = ((const float4*)down)[i];
        bf16x4 b;
        b[0] = (__bf16)d.x; b[1] = (__bf16)d.y; b[2] = (__bf16)d.z; b[3] = (__bf16)d.w;
        ((bf16x4*)downb)[i] = b;
    }
}

// ------------------------------------------- LayerNorm + gating + top-2 route
// One wave32 per token. H=1024 -> 32 floats per lane.
__global__ __launch_bounds__(256) void k_ln_gate(const float* __restrict__ x,
                                                 const float* __restrict__ gate_w,
                                                 const float* __restrict__ ln_g,
                                                 const float* __restrict__ ln_b,
                                                 __bf16* __restrict__ xnb,
                                                 int* __restrict__ tokl,
                                                 float* __restrict__ wl,
                                                 int* __restrict__ cnt,
                                                 float* __restrict__ probsum) {
    int lane = threadIdx.x & 31;
    int wave = threadIdx.x >> 5;
    int t = blockIdx.x * 8 + wave;
    if (t >= TT) return;

    float v[32];
    const float4* x4 = (const float4*)(x + (size_t)t * H);
#pragma unroll
    for (int j = 0; j < 8; ++j) {
        float4 q = x4[j * 32 + lane];
        v[4*j+0] = q.x; v[4*j+1] = q.y; v[4*j+2] = q.z; v[4*j+3] = q.w;
    }
    float s = 0.0f;
#pragma unroll
    for (int i = 0; i < 32; ++i) s += v[i];
#pragma unroll
    for (int m = 16; m >= 1; m >>= 1) s += __shfl_xor(s, m, 32);
    float mu = s * (1.0f / (float)H);
    float s2 = 0.0f;
#pragma unroll
    for (int i = 0; i < 32; ++i) { float d = v[i] - mu; s2 += d * d; }
#pragma unroll
    for (int m = 16; m >= 1; m >>= 1) s2 += __shfl_xor(s2, m, 32);
    float rstd = rsqrtf(s2 * (1.0f / (float)H) + 1e-5f);

    const float4* g4 = (const float4*)ln_g;
    const float4* b4 = (const float4*)ln_b;
#pragma unroll
    for (int j = 0; j < 8; ++j) {
        float4 g = g4[j * 32 + lane];
        float4 b = b4[j * 32 + lane];
        v[4*j+0] = (v[4*j+0] - mu) * rstd * g.x + b.x;
        v[4*j+1] = (v[4*j+1] - mu) * rstd * g.y + b.y;
        v[4*j+2] = (v[4*j+2] - mu) * rstd * g.z + b.z;
        v[4*j+3] = (v[4*j+3] - mu) * rstd * g.w + b.w;
    }
    // store normalized activations as bf16
    __bf16* xr = xnb + (size_t)t * H;
#pragma unroll
    for (int j = 0; j < 8; ++j) {
        bf16x4 o;
        o[0] = (__bf16)v[4*j+0]; o[1] = (__bf16)v[4*j+1];
        o[2] = (__bf16)v[4*j+2]; o[3] = (__bf16)v[4*j+3];
        *(bf16x4*)(xr + j * 128 + lane * 4) = o;
    }
    // gate scores (8 dots of length 1024)
    const float4* gw4 = (const float4*)gate_w;
    float sc[E];
#pragma unroll
    for (int e = 0; e < E; ++e) {
        float a = 0.0f;
#pragma unroll
        for (int j = 0; j < 8; ++j) {
            float4 g = gw4[e * 256 + j * 32 + lane];
            a += g.x * v[4*j+0] + g.y * v[4*j+1] + g.z * v[4*j+2] + g.w * v[4*j+3];
        }
        sc[e] = a;
    }
#pragma unroll
    for (int e = 0; e < E; ++e)
#pragma unroll
        for (int m = 16; m >= 1; m >>= 1) sc[e] += __shfl_xor(sc[e], m, 32);

    if (lane == 0) {
        float mx = sc[0];
#pragma unroll
        for (int e = 1; e < E; ++e) mx = fmaxf(mx, sc[e]);
        float p[E], den = 0.0f;
#pragma unroll
        for (int e = 0; e < E; ++e) { p[e] = __expf(sc[e] - mx); den += p[e]; }
        float rden = 1.0f / den;
#pragma unroll
        for (int e = 0; e < E; ++e) p[e] *= rden;
        // top-2 (lowest index wins ties, like lax.top_k)
        int i1 = 0;
#pragma unroll
        for (int e = 1; e < E; ++e) if (p[e] > p[i1]) i1 = e;
        int i2 = (i1 == 0) ? 1 : 0;
#pragma unroll
        for (int e = 0; e < E; ++e) if (e != i1 && p[e] > p[i2]) i2 = e;
        float w1 = p[i1], w2 = p[i2];
        float rw = 1.0f / (w1 + w2);
        w1 *= rw; w2 *= rw;
        int p1 = atomicAdd(&cnt[i1], 1);
        tokl[i1 * TT + p1] = t; wl[i1 * TT + p1] = w1;
        int p2 = atomicAdd(&cnt[i2], 1);
        tokl[i2 * TT + p2] = t; wl[i2 * TT + p2] = w2;
#pragma unroll
        for (int e = 0; e < E; ++e) atomicAdd(&probsum[e], p[e]);
    }
}

// --------------------------------------------- fused grouped FFN (WMMA bf16)
// One workgroup = 32 routed tokens of one expert x one half of the hidden dim.
// LDS: 64 KB x_norm tile + 128 KB h tile (CDNA5 WGP has 320 KB).
// Each B fragment feeds 2 WMMAs (2 m-tiles) -> 1.5 b128 loads per WMMA.
__global__ __launch_bounds__(512) void k_moe_ffn(const __bf16* __restrict__ wupb,
                                                 const __bf16* __restrict__ wdownb,
                                                 const float* __restrict__ up_b,
                                                 const float* __restrict__ down_b,
                                                 const __bf16* __restrict__ xnb,
                                                 const int* __restrict__ tokl,
                                                 const float* __restrict__ wl,
                                                 const int* __restrict__ cnt,
                                                 float* __restrict__ out) {
    __shared__ __bf16 xnS[ROWS * H];        // 64 KB
    __shared__ __bf16 hS[ROWS * FH];        // 128 KB
    __shared__ int   sTok[ROWS];
    __shared__ float sW[ROWS];

    int e   = blockIdx.x >> 9;              // 512 blocks per expert
    int idx = blockIdx.x & 511;
    int rb  = idx >> 1;                     // row-block (32 tokens)
    int fh  = idx & 1;                      // which half of F
    int base = rb * ROWS;
    int n_e = cnt[e];
    if (base >= n_e) return;

    int tid = threadIdx.x;
    int lane = tid & 31;
    int wave = tid >> 5;                    // 16 waves

    if (tid < ROWS) {
        int r = base + tid;
        if (r < n_e) { sTok[tid] = tokl[e * TT + r]; sW[tid] = wl[e * TT + r]; }
        else         { sTok[tid] = tokl[e * TT + base]; sW[tid] = 0.0f; }
    }
    __syncthreads();

    // stage 32 x_norm rows into LDS (b128 copies)
    {
        const uint4* xsrc = (const uint4*)xnb;   // 128 uint4 per row
        uint4* xdst = (uint4*)xnS;
#pragma unroll
        for (int i = tid; i < ROWS * 128; i += 512) {
            int r = i >> 7, c = i & 127;
            xdst[r * 128 + c] = xsrc[(size_t)sTok[r] * 128 + c];
        }
    }
    __syncthreads();

    const int mrow  = lane & 15;            // row/col index within a tile
    const int offA  = (lane >> 4) << 4;     // A-fragment: lanes 16-31 take K+16
    const int offB  = (lane >> 4) << 4;     // B-fragment: lanes 16-31 take K+16
    const int row16 = (lane >> 4) << 3;     // C/D: lanes 16-31 hold M+8

    // ---- GEMM1: h[32,FH] = gelu(xn[32,H] @ Wup[e][fh]^T + up_b) -----------
    {
        const __bf16* xrow0 = xnS + (size_t)mrow * H;
        const __bf16* xrow1 = xnS + (size_t)(16 + mrow) * H;
        const __bf16* wu = wupb + (size_t)e * F * H;
#pragma unroll 1
        for (int g = 0; g < 2; ++g) {       // 2 groups of 4 n-tiles per wave
            int nloc = wave * 128 + g * 64; // local column base in [0,FH)
            f32x8 acc[2][4] = {};
            const __bf16* bp[4];
#pragma unroll
            for (int tI = 0; tI < 4; ++tI) {
                bp[tI] = wu + (size_t)(fh * FH + nloc + tI * 16 + mrow) * H + offB;
                __builtin_prefetch(bp[tI], 0, 1);      // global_prefetch_b8
            }
#pragma unroll 4
            for (int k0 = 0; k0 < H; k0 += 32) {
                bf16x16 a0 = frag_a(xrow0, k0, offA);
                bf16x16 a1 = frag_a(xrow1, k0, offA);
#pragma unroll
                for (int tI = 0; tI < 4; ++tI) {
                    bf16x16 b = *(const bf16x16*)(bp[tI] + k0);
                    acc[0][tI] = wmma_bf16(a0, b, acc[0][tI]);
                    acc[1][tI] = wmma_bf16(a1, b, acc[1][tI]);
                }
            }
#pragma unroll
            for (int tI = 0; tI < 4; ++tI) {
                int lcol = nloc + tI * 16 + mrow;
                float bias = up_b[e * F + fh * FH + lcol];
#pragma unroll
                for (int mt = 0; mt < 2; ++mt) {
#pragma unroll
                    for (int j = 0; j < 8; ++j) {
                        float xv = acc[mt][tI][j] + bias;
                        float gl = 0.5f * xv * (1.0f + erff(xv * 0.70710678118f));
                        hS[(mt * 16 + j + row16) * FH + lcol] = (__bf16)gl;
                    }
                }
            }
        }
    }
    __syncthreads();

    // ---- GEMM2: y[32,H] += h[32,FH] @ Wdown[e][:,fh]^T (+bias on fh==0) ---
    {
        const __bf16* hrow0 = hS + (size_t)mrow * FH;
        const __bf16* hrow1 = hS + (size_t)(16 + mrow) * FH;
        const __bf16* wd = wdownb + (size_t)e * H * F;
        int nb = wave * 64;                  // 16 waves x 64 cols = 1024
        f32x8 acc[2][4] = {};
        const __bf16* bp[4];
#pragma unroll
        for (int tI = 0; tI < 4; ++tI) {
            bp[tI] = wd + (size_t)(nb + tI * 16 + mrow) * F + fh * FH + offB;
            __builtin_prefetch(bp[tI], 0, 1);
        }
#pragma unroll 4
        for (int k0 = 0; k0 < FH; k0 += 32) {
            bf16x16 a0 = frag_a(hrow0, k0, offA);
            bf16x16 a1 = frag_a(hrow1, k0, offA);
#pragma unroll
            for (int tI = 0; tI < 4; ++tI) {
                bf16x16 b = *(const bf16x16*)(bp[tI] + k0);
                acc[0][tI] = wmma_bf16(a0, b, acc[0][tI]);
                acc[1][tI] = wmma_bf16(a1, b, acc[1][tI]);
            }
        }
#pragma unroll
        for (int tI = 0; tI < 4; ++tI) {
            int col = nb + tI * 16 + mrow;
            float bias = (fh == 0) ? down_b[e * H + col] : 0.0f;
#pragma unroll
            for (int mt = 0; mt < 2; ++mt) {
#pragma unroll
                for (int j = 0; j < 8; ++j) {
                    int M = mt * 16 + j + row16;
                    float y = (acc[mt][tI][j] + bias) * sW[M];
                    unsafeAtomicAdd(&out[(size_t)sTok[M] * H + col], y);
                }
            }
        }
    }
}

// ------------------------------------------------------------- LB loss scalar
__global__ void k_finalize(const int* __restrict__ cnt,
                           const float* __restrict__ probsum,
                           float* __restrict__ out_loss) {
    if (threadIdx.x == 0) {
        float lb = 0.0f;
        for (int e = 0; e < E; ++e) {
            float freq = (float)cnt[e] / (float)(TT * TK);
            float pr = probsum[e] / (float)TT;
            lb += freq * pr;
        }
        *out_loss = 0.01f * (float)E * lb;
    }
}

extern "C" void kernel_launch(void* const* d_in, const int* in_sizes, int n_in,
                              void* d_out, int out_size, void* d_ws, size_t ws_size,
                              hipStream_t stream) {
    const float* x      = (const float*)d_in[0];
    const float* gate_w = (const float*)d_in[1];
    const float* up_w   = (const float*)d_in[2];
    const float* up_b   = (const float*)d_in[3];
    const float* down_w = (const float*)d_in[4];
    const float* down_b = (const float*)d_in[5];
    const float* ln_g   = (const float*)d_in[6];
    const float* ln_b   = (const float*)d_in[7];
    float* out = (float*)d_out;

    char* ws = (char*)d_ws;
    size_t o = 0;
    __bf16* wupb   = (__bf16*)(ws + o); o += (size_t)E * F * H * 2;   // 64 MB
    __bf16* wdownb = (__bf16*)(ws + o); o += (size_t)E * F * H * 2;   // 64 MB
    __bf16* xnb    = (__bf16*)(ws + o); o += (size_t)TT * H * 2;      // 16 MB
    int*    tokl   = (int*)  (ws + o);  o += (size_t)E * TT * 4;
    float*  wlst   = (float*)(ws + o);  o += (size_t)E * TT * 4;
    int*    cnt    = (int*)  (ws + o);  o += 256;
    float*  probs  = (float*)(ws + o);  o += 256;

    k_zero<<<2048, 256, 0, stream>>>(out, cnt, probs);
    k_convert<<<8192, 256, 0, stream>>>(up_w, down_w, wupb, wdownb);
    k_ln_gate<<<TT / 8, 256, 0, stream>>>(x, gate_w, ln_g, ln_b, xnb, tokl, wlst, cnt, probs);
    k_moe_ffn<<<E * 512, 512, 0, stream>>>(wupb, wdownb, up_b, down_b, xnb, tokl, wlst, cnt, out);
    k_finalize<<<1, 32, 0, stream>>>(cnt, probs, out + (size_t)TT * H);
}
